// CustomLSTMImpl_69879117906366
// MI455X (gfx1250) — matrix operand order repaired and checked by reading, not compiled, not run on hardware
//
#include <hip/hip_runtime.h>

typedef _Float16 half_t;
typedef __attribute__((ext_vector_type(16))) _Float16 v16h;
typedef __attribute__((ext_vector_type(8)))  _Float16 v8h;
typedef __attribute__((ext_vector_type(8)))  float    v8f;

#define TT  512
#define BB  64
#define II  128
#define HH  256
#define LLL 6
#define G4H 1024   // 4*H
#define LDW 264    // padded LDS row stride in halves (528 B -> conflict-free)

__device__ __forceinline__ float sigf(float x) {
  return 1.0f / (1.0f + __expf(-x));
}
__device__ __forceinline__ float tanh_fast(float x) {
  float xc = fminf(10.0f, fmaxf(-10.0f, x));
  float e2 = __expf(2.0f * xc);
  return (e2 - 1.0f) / (e2 + 1.0f);
}

__device__ __forceinline__ v16h combine8(v8h c0, v8h c1) {
  v16h r;
#pragma unroll
  for (int i = 0; i < 8; ++i) { r[i] = c0[i]; r[i + 8] = c1[i]; }
  return r;
}

// A-matrix fragment 16x32 f16 (ISA 7.12.2). p is the lane-adjusted base:
//   p = rowbase + (lane&15)*ld + 8*(lane>>4).  off in halves (compile-time).
__device__ __forceinline__ v16h frag_a(const half_t* p, int off) {
  return combine8(*(const v8h*)(p + off), *(const v8h*)(p + off + 16));
}

// B-matrix fragment 32x16 f16. p is the lane-adjusted base:
//   p = W + (lane&15)*ld + 8*(lane>>4).  off in halves (compile-time).
__device__ __forceinline__ v16h frag_b(const half_t* p, int off) {
  return combine8(*(const v8h*)(p + off), *(const v8h*)(p + off + 16));
}

// One row-half (2 row-tiles) of a projection: acc += A(32xK) @ B(Kx64).
// aB/bB are lane-adjusted bases; all displacements are literals.
template <int LDA, int WSTR, int NKS>
__device__ __forceinline__ void gemm_half(v8f (&acc)[2][4],
                                          const half_t* aB, const half_t* bB) {
#pragma unroll
  for (int ks = 0; ks < NKS; ++ks) {
    const int k0 = ks * 32;
    v16h b[4];
#pragma unroll
    for (int g = 0; g < 4; ++g) b[g] = frag_b(bB, g * WSTR + k0);
#pragma unroll
    for (int r = 0; r < 2; ++r) {
      v16h a = frag_a(aB, r * 16 * LDA + k0);
#pragma unroll
      for (int g = 0; g < 4; ++g)
        acc[r][g] = __builtin_amdgcn_wmma_f32_16x16x32_f16(
            false, a, false, b[g], (short)0, acc[r][g], false, false);
    }
  }
}

// Pointwise LSTM cell update for one row-half; hb/yb pre-offset to the half.
__device__ __forceinline__ void point_half(v8f (&acc)[2][4], float (&cst)[2][8],
                                           half_t* hb, half_t* yb) {
#pragma unroll
  for (int r = 0; r < 2; ++r)
#pragma unroll
    for (int v = 0; v < 8; ++v) {
      const float ig = sigf(acc[r][0][v]);
      const float fg = sigf(acc[r][1][v]);
      const float gg = tanh_fast(acc[r][2][v]);
      const float og = sigf(acc[r][3][v]);
      const float c  = fg * cst[r][v] + ig * gg;
      cst[r][v] = c;
      const half_t hh = (half_t)(og * tanh_fast(c));
      hb[(16 * r + v) * LDW] = hh;
      yb[(16 * r + v) * HH]  = hh;
    }
}

__global__ void flags_init(unsigned int* flags) {
  if (threadIdx.x < 16) flags[threadIdx.x] = 0u;
}

// ---------------------------------------------------------------------------
// One LSTM layer, run by one 512-thread workgroup (16 waves).
// Wave w owns j-columns [16w,16w+16) for all 4 row-tiles and all 4 gates.
// Each step is split into two row-half passes so only 64 accumulator VGPRs
// are live at a time. A laundered zero offset is added to the weight bases
// each iteration so LICM cannot hoist (and then spill) the loop-invariant
// weight fragment loads -- they must stream from cache; keeping the offset
// (not the pointer) opaque preserves the global address space, so these
// stay global_load_b128 (LOADcnt only, no flat/LDS coupling).
// ---------------------------------------------------------------------------
template <int KIN>
__device__ __forceinline__ void run_layer(
    const int l, const int tid, const int lo, const int hi, const int j,
    const half_t* __restrict__ inp, const half_t* __restrict__ wih,
    const half_t* __restrict__ whh, const half_t* __restrict__ bih,
    const half_t* __restrict__ bhh, const half_t* __restrict__ h0,
    const half_t* __restrict__ c0, half_t* __restrict__ ys,
    half_t* __restrict__ hn, half_t* __restrict__ cn,
    unsigned int* __restrict__ progress, half_t* hbuf) {

  // Warm caches with this layer's weights (global_prefetch_b8).
  for (int off = tid * 64; off < G4H * HH; off += 512 * 64)
    __builtin_prefetch(whh + off, 0, 3);
  for (int off = tid * 64; off < G4H * KIN; off += 512 * 64)
    __builtin_prefetch(wih + off, 0, 3);

  // Lane-adjusted weight bases; per-gate/per-kstep displacements are literals.
  const half_t* whb = whh + (size_t)j * HH + 8 * hi;
  const half_t* wib = wih + (size_t)j * KIN + 8 * hi;

  // h0 -> hbuf[0] (cooperative, vectorized, padded rows).
#pragma unroll
  for (int idx = tid * 8; idx < BB * HH; idx += 512 * 8) {
    const int row = idx / HH, col = idx % HH;
    *(v8h*)(hbuf + row * LDW + col) = *(const v8h*)(h0 + idx);
  }

  float biasr[4];
#pragma unroll
  for (int g = 0; g < 4; ++g)
    biasr[g] = (float)bih[256 * g + j] + (float)bhh[256 * g + j];

  // c state, split per row-half so all indexing stays compile-time.
  float cst0[2][8], cst1[2][8];
#pragma unroll
  for (int r = 0; r < 2; ++r)
#pragma unroll
    for (int v = 0; v < 8; ++v) {
      cst0[r][v] = (float)c0[(16 * r + 8 * hi + v) * HH + j];
      cst1[r][v] = (float)c0[(32 + 16 * r + 8 * hi + v) * HH + j];
    }
  __syncthreads();

  int cur = 0;
  for (int t = 0; t < TT; ++t) {
    // Laundered zero offset: value is 0, but the compiler must assume it
    // changes every iteration -> weight loads cannot be hoisted out of the
    // t-loop, yet the bases keep global address-space provenance.
    int lz = 0;
    asm volatile("" : "+v"(lz));
    const half_t* whb_t = whb + lz;
    const half_t* wib_t = wib + lz;

    const half_t* hA = hbuf + cur * (BB * LDW) + lo * LDW + 8 * hi;
    const half_t* xb = inp + ((size_t)t * BB + lo) * KIN + 8 * hi;
    half_t* hb = hbuf + (cur ^ 1) * (BB * LDW) + 8 * hi * LDW + j;
    half_t* yb = ys + ((size_t)t * BB + 8 * hi) * HH + j;

    // ================= pass 0: rows 0..31 =================
    {
      v8f acc[2][4];
#pragma unroll
      for (int r = 0; r < 2; ++r)
#pragma unroll
        for (int g = 0; g < 4; ++g)
#pragma unroll
          for (int v = 0; v < 8; ++v) acc[r][g][v] = biasr[g];

      // Recurrent half first: overlaps the producer's tail.
      gemm_half<LDW, 256 * HH, 8>(acc, hA, whb_t);

      // Acquire: previous layer must have published step t.
      if (l > 0) {
        if (tid == 0) {
          while (__hip_atomic_load(&progress[l - 1], __ATOMIC_ACQUIRE,
                                   __HIP_MEMORY_SCOPE_AGENT) < (unsigned)(t + 1)) {
            __builtin_amdgcn_s_sleep(1);
          }
          __threadfence();   // invalidate WGP caches before reading inp
        }
        __syncthreads();
      }

      gemm_half<KIN, 256 * KIN, KIN / 32>(acc, xb, wib_t);
      point_half(acc, cst0, hb, yb);
    }

    // ================= pass 1: rows 32..63 =================
    {
      v8f acc[2][4];
#pragma unroll
      for (int r = 0; r < 2; ++r)
#pragma unroll
        for (int g = 0; g < 4; ++g)
#pragma unroll
          for (int v = 0; v < 8; ++v) acc[r][g][v] = biasr[g];

      gemm_half<LDW, 256 * HH, 8>(acc, hA + 32 * LDW, whb_t);
      gemm_half<KIN, 256 * KIN, KIN / 32>(acc, xb + 32 * KIN, wib_t);
      point_half(acc, cst1, hb + 32 * LDW, yb + 32 * HH);
    }

    // ---- Release: flush stores (all threads), barrier, publish ----
    if (l < LLL - 1) {
      __threadfence();
      __syncthreads();
      if (tid == 0)
        __hip_atomic_store(&progress[l], (unsigned)(t + 1), __ATOMIC_RELEASE,
                           __HIP_MEMORY_SCOPE_AGENT);
    } else {
      __syncthreads();   // hbuf handoff only
    }
    cur ^= 1;
  }

  // Final h_n / c_n for this layer.
  {
    const half_t* hc = hbuf + cur * (BB * LDW);
#pragma unroll
    for (int r = 0; r < 2; ++r)
#pragma unroll
      for (int v = 0; v < 8; ++v) {
        const int m0 = 16 * r + 8 * hi + v;
        const int m1 = 32 + m0;
        hn[m0 * HH + j] = hc[m0 * LDW + j];
        cn[m0 * HH + j] = (half_t)cst0[r][v];
        hn[m1 * HH + j] = hc[m1 * LDW + j];
        cn[m1 * HH + j] = (half_t)cst1[r][v];
      }
  }
}

// ---------------------------------------------------------------------------
// Persistent fused LSTM pipeline: one workgroup per layer (grid = 6).
// ---------------------------------------------------------------------------
__global__ void __launch_bounds__(512)
lstm_fused(const half_t* __restrict__ x,
           const half_t* __restrict__ wih0,
           const half_t* __restrict__ wihr,
           const half_t* __restrict__ whh_all,
           const half_t* __restrict__ bih_all,
           const half_t* __restrict__ bhh_all,
           const half_t* __restrict__ h0_all,
           const half_t* __restrict__ c0_all,
           half_t* __restrict__ ysbufs,      // 5 x [T*B, H] inter-layer buffers
           half_t* __restrict__ out,         // [T*B, H] final layer output
           half_t* __restrict__ hn_base,     // [L, B, H]
           half_t* __restrict__ cn_base,     // [L, B, H]
           unsigned int* __restrict__ progress) {
  __shared__ half_t hbuf[2 * BB * LDW];      // ~66 KB double buffer, padded rows

  const int l    = blockIdx.x;
  const int tid  = threadIdx.x;
  const int w    = tid >> 5;                 // wave id = j-tile (0..15)
  const int lane = tid & 31;
  const int lo   = lane & 15, hi = lane >> 4;
  const int j    = 16 * w + lo;              // this lane's h column

  const half_t* whh = whh_all + (size_t)l * G4H * HH;
  const half_t* bih = bih_all + (size_t)l * G4H;
  const half_t* bhh = bhh_all + (size_t)l * G4H;
  const half_t* h0  = h0_all + (size_t)l * BB * HH;
  const half_t* c0  = c0_all + (size_t)l * BB * HH;
  half_t* ys = (l == LLL - 1) ? out : (ysbufs + (size_t)l * TT * BB * HH);
  half_t* hn = hn_base + (size_t)l * BB * HH;
  half_t* cn = cn_base + (size_t)l * BB * HH;

  if (l == 0) {
    run_layer<II>(l, tid, lo, hi, j, x, wih0, whh, bih, bhh, h0, c0,
                  ys, hn, cn, progress, hbuf);
  } else {
    const half_t* inp = ysbufs + (size_t)(l - 1) * TT * BB * HH;
    const half_t* wih = wihr + (size_t)(l - 1) * G4H * HH;
    run_layer<HH>(l, tid, lo, hi, j, inp, wih, whh, bih, bhh, h0, c0,
                  ys, hn, cn, progress, hbuf);
  }
}

// ---------------------------------------------------------------------------
// Host side: zero progress flags (tiny kernel, graph-capture safe), then
// launch the 6-workgroup persistent pipeline.
// ---------------------------------------------------------------------------
extern "C" void kernel_launch(void* const* d_in, const int* in_sizes, int n_in,
                              void* d_out, int out_size, void* d_ws, size_t ws_size,
                              hipStream_t stream) {
  (void)in_sizes; (void)n_in; (void)out_size; (void)ws_size;

  const half_t* x    = (const half_t*)d_in[0];
  const half_t* h0   = (const half_t*)d_in[1];
  const half_t* c0   = (const half_t*)d_in[2];
  const half_t* wih0 = (const half_t*)d_in[3];
  const half_t* wihr = (const half_t*)d_in[4];
  const half_t* whh  = (const half_t*)d_in[5];
  const half_t* bih  = (const half_t*)d_in[6];
  const half_t* bhh  = (const half_t*)d_in[7];
  half_t* out = (half_t*)d_out;

  // Workspace: 5 inter-layer activation buffers (16 MB each) + flags.
  half_t* ysbufs = (half_t*)d_ws;
  unsigned int* flags =
      (unsigned int*)((char*)d_ws + (size_t)(LLL - 1) * TT * BB * HH * sizeof(half_t));

  half_t* hn_base = out + (size_t)TT * BB * HH;
  half_t* cn_base = hn_base + (size_t)LLL * BB * HH;

  flags_init<<<1, 32, 0, stream>>>(flags);
  lstm_fused<<<dim3(LLL), 512, 0, stream>>>(x, wih0, wihr, whh, bih, bhh, h0, c0,
                                            ysbufs, out, hn_base, cn_base, flags);
}
